// BackdoorAdjustedAttention_52080773431667
// MI455X (gfx1250) — matrix operand order repaired and checked by reading, not compiled, not run on hardware
//
#include <hip/hip_runtime.h>
#include <cstdint>
#include <cstddef>

// ---------------------------------------------------------------------------
// BackdoorAdjustedAttention on MI455X (gfx1250, wave32, WMMA).
// bf16 -> f32 accumulate via v_wmma_f32_16x16x32_bf16 everywhere.
// Double-buffered async global->LDS pipelines (ASYNCcnt), ONE barrier per
// tile step: wait(own async) -> barrier -> prefetch next buffer -> compute.
// ---------------------------------------------------------------------------

typedef unsigned short u16;
typedef __attribute__((ext_vector_type(16))) __bf16 v16bf;
typedef __attribute__((ext_vector_type(8)))  float  v8f;

#define DEVINL __device__ __forceinline__

static constexpr int   B_   = 4;
static constexpr int   N_   = 2048;
static constexpr int   D_   = 1024;
static constexpr int   H_   = 16;
static constexpr int   DK_  = 64;
static constexpr int   M_   = B_ * N_;       // 8192 flattened rows
static constexpr float SCALE_ = 0.125f;      // 1/sqrt(64)

DEVINL u16 f2bf(float f) {
  unsigned u = __float_as_uint(f);
  u += 0x7FFFu + ((u >> 16) & 1u);           // round-to-nearest-even
  return (u16)(u >> 16);
}
DEVINL float bf2f(u16 h) { return __uint_as_float(((unsigned)h) << 16); }

union FragCast { struct { uint4 a, b; } q; v16bf v; };

// Build a 16-element bf16 fragment from two 16-byte chunks.
DEVINL v16bf ldfrag(const u16* lo, const u16* hi) {
  FragCast u;
  u.q.a = *reinterpret_cast<const uint4*>(lo);
  u.q.b = *reinterpret_cast<const uint4*>(hi);
  return u.v;
}

// Gather a 16-element bf16 B fragment down a column (table k order).
DEVINL v16bf ldfragT(const u16* col, int stride, int hf) {
  union { u16 e[16]; v16bf v; } u;
#pragma unroll
  for (int s = 0; s < 16; ++s) {
    int k = (s < 8) ? (8 * hf + s) : (16 + 8 * hf + (s - 8));
    u.e[s] = col[k * stride];
  }
  return u.v;
}

DEVINL v8f wmma_bf16(v16bf a, v16bf b, v8f c) {
  return __builtin_amdgcn_wmma_f32_16x16x32_bf16(
      /*neg_a=*/false, a, /*neg_b=*/false, b,
      /*c_mod=*/(short)0, c, /*reuse_a=*/false, /*reuse_b=*/false);
}

// --- CDNA5 async global->LDS (tracked by ASYNCcnt) -------------------------
typedef __attribute__((address_space(3))) void lds_void;

DEVINL unsigned lds_off(void* p) {           // 32-bit LDS byte offset
  return (unsigned)(uintptr_t)(lds_void*)p;
}
DEVINL void async_ld16(unsigned lds, const u16* g) {
  asm volatile("global_load_async_to_lds_b128 %0, %1, off"
               :: "v"(lds), "v"((unsigned long long)(uintptr_t)g)
               : "memory");
}
DEVINL void wait_async0() {
  asm volatile("s_wait_asynccnt 0x0" ::: "memory");
}

// ---------------------------------------------------------------------------
// fp32 -> bf16 elementwise conversion (grid-stride)
// ---------------------------------------------------------------------------
__global__ void k_f2bf(const float* __restrict__ in, u16* __restrict__ out,
                       long long n) {
  long long i = (long long)blockIdx.x * blockDim.x + threadIdx.x;
  long long stride = (long long)gridDim.x * blockDim.x;
  for (; i < n; i += stride) out[i] = f2bf(in[i]);
}

// ---------------------------------------------------------------------------
// fp32 -> bf16 transpose (n x n), 32x32 LDS tiles
// ---------------------------------------------------------------------------
__global__ __launch_bounds__(256)
void k_transpose_f2bf(const float* __restrict__ in, u16* __restrict__ out, int n) {
  __shared__ float tile[32][33];
  int bx = blockIdx.x * 32, by = blockIdx.y * 32;
  int tx = threadIdx.x & 31, ty = threadIdx.x >> 5;     // 8 row-passes
  for (int r = ty; r < 32; r += 8)
    tile[r][tx] = in[(size_t)(by + r) * n + (bx + tx)];
  __syncthreads();
  for (int r = ty; r < 32; r += 8)
    out[(size_t)(bx + r) * n + (by + tx)] = f2bf(tile[tx][r]);
}

// ---------------------------------------------------------------------------
// Generic bf16 GEMM:  C[M,Nc] = A[M,Kd] @ Bm[Kd,Nc] (+ bias[Nc])
//                                          (+ extra[row & extraMask, Nc])
// WG tile 128x256, 8 waves (2 x 4), wave tile 64x64 (4x4 WMMA), K-step 32.
// Double-buffered async->LDS pipeline, one barrier per K-step.
// ---------------------------------------------------------------------------
__global__ __launch_bounds__(256)
void k_gemm(const u16* __restrict__ A, const u16* __restrict__ Bm,
            const float* __restrict__ bias, const u16* __restrict__ extra,
            int extraMask, float* __restrict__ outF, u16* __restrict__ outB,
            int M, int Nc, int Kd) {
  constexpr int ASTR = 40;   // (32+8) bf16 -> 80B rows, 16B aligned
  constexpr int BSTR = 264;  // (256+8) bf16 -> 528B rows, 16B aligned
  constexpr int ASZ  = 128 * ASTR;
  constexpr int BSZ  = 32 * BSTR;
  __shared__ __align__(16) u16 As[2 * ASZ];
  __shared__ __align__(16) u16 Bs[2 * BSZ];

  const int t    = threadIdx.x;
  const int wave = t >> 5;
  const int lane = t & 31;
  const int m0   = blockIdx.x * 128;
  const int n0   = blockIdx.y * 256;
  const int wm   = (wave & 1) * 64;
  const int wn   = (wave >> 1) * 64;
  const int mrow = lane & 15;       // A-fragment M index
  const int hf   = lane >> 4;       // half-wave select

  const int nk = Kd >> 5;
  v8f acc[4][4] = {};

  auto stage = [&](int it, int buf) {
    const int kk = it << 5;
    u16* Asb = As + buf * ASZ;
    u16* Bsb = Bs + buf * BSZ;
    {  // A tile 128x32 : 2 async 16B transfers per thread
      int row = t >> 1, seg = t & 1;
      const u16* g = &A[(size_t)(m0 + row) * Kd + kk + seg * 16];
      unsigned l = lds_off(&Asb[row * ASTR + seg * 16]);
      async_ld16(l,      g);
      async_ld16(l + 16, g + 8);
    }
    {  // B tile 32x256 : 4 async 16B transfers per thread
      int row = t >> 3, seg = t & 7;
      const u16* g = &Bm[(size_t)(kk + row) * Nc + n0 + seg * 32];
      unsigned l = lds_off(&Bsb[row * BSTR + seg * 32]);
#pragma unroll
      for (int c = 0; c < 4; ++c) async_ld16(l + 16 * c, g + 8 * c);
    }
  };

  stage(0, 0);
  for (int it = 0; it < nk; ++it) {
    const int buf = it & 1;
    wait_async0();        // my tile-it transfers have landed
    __syncthreads();      // everyone's landed; everyone done reading buf^1
    if (it + 1 < nk) stage(it + 1, buf ^ 1);   // overlaps with compute below

    const u16* Asb = As + buf * ASZ;
    const u16* Bsb = Bs + buf * BSZ;
    v16bf af[4];
#pragma unroll
    for (int mt = 0; mt < 4; ++mt) {
      const u16* base = &Asb[(wm + mt * 16 + mrow) * ASTR];
      af[mt] = ldfrag(base + 8 * hf, base + 16 + 8 * hf);
    }
#pragma unroll
    for (int nt = 0; nt < 4; ++nt) {
      const u16* base = &Bsb[wn + nt * 16 + mrow]; // lane&15 = N col
      v16bf bfm = ldfragT(base, BSTR, hf);
#pragma unroll
      for (int mt = 0; mt < 4; ++mt)
        acc[mt][nt] = wmma_bf16(af[mt], bfm, acc[mt][nt]);
    }
  }

  // Epilogue: C layout -> element (m = v + 8*hf, n = lane&15)
  const int ncol = lane & 15;
#pragma unroll
  for (int mt = 0; mt < 4; ++mt)
#pragma unroll
    for (int nt = 0; nt < 4; ++nt)
      for (int v = 0; v < 8; ++v) {
        int row = m0 + wm + mt * 16 + v + 8 * hf;
        int col = n0 + wn + nt * 16 + ncol;
        float x = acc[mt][nt][v];
        if (bias)  x += bias[col];
        if (extra) x += bf2f(extra[(size_t)(row & extraMask) * Nc + col]);
        if (outF)  outF[(size_t)row * Nc + col] = x;
        if (outB)  outB[(size_t)row * Nc + col] = f2bf(x);
      }
}

// ---------------------------------------------------------------------------
// Flash-style attention with multiplicative pre-softmax backdoor mask.
// grid = (N/128, B*H); block = 256 (8 waves); wave handles 16 query rows.
// Double-buffered staging: V via async->LDS; K loaded to VGPRs early and
// ds_stored transposed after compute. One barrier per j-chunk.
// ---------------------------------------------------------------------------
__global__ __launch_bounds__(256)
void k_attn(const u16* __restrict__ Qb, const u16* __restrict__ Kb,
            const u16* __restrict__ Vb, const float* __restrict__ mask,
            u16* __restrict__ Ob) {
  constexpr int KSTR = 40;  // 64 x (32+8)  -> 80B rows
  constexpr int VSTR = 72;  // 32 x (64+8)  -> 144B rows
  constexpr int PSTR = 40;  // 16 x (32+8)  -> 80B rows
  constexpr int KSZ  = 64 * KSTR;
  constexpr int VSZ  = 32 * VSTR;
  __shared__ __align__(16) u16 Kst[2 * KSZ];         // [dk][j], double buffer
  __shared__ __align__(16) u16 Vs[2 * VSZ];          // [j][dk], double buffer
  __shared__ __align__(16) u16 Ps[8][16 * PSTR];     // per-wave P staging

  const int t    = threadIdx.x;
  const int wave = t >> 5;
  const int lane = t & 31;
  const int b    = blockIdx.y >> 4;        // H_ = 16
  const int h    = blockIdx.y & 15;
  const int i0   = blockIdx.x * 128 + wave * 16;
  const int mrow = lane & 15;
  const int hf   = lane >> 4;
  const int jr   = t >> 3;                 // staging: key row 0..31
  const int seg  = t & 7;                  // staging: dk octet

  // Q fragments for this wave's 16 rows x 64 dk (2 K-steps of 32), kept live.
  v16bf qa[2];
  {
    const u16* p = Qb + (size_t)(b * N_ + i0 + mrow) * D_ + h * DK_;
    qa[0] = ldfrag(p + 8 * hf,      p + 16 + 8 * hf);
    qa[1] = ldfrag(p + 32 + 8 * hf, p + 48 + 8 * hf);
  }

  v8f o[4] = {};
  float mrw[8], lrw[8];
#pragma unroll
  for (int v = 0; v < 8; ++v) { mrw[v] = -1e30f; lrw[v] = 0.0f; }

  auto stageV = [&](int jt, int buf) {
    async_ld16(lds_off(&Vs[buf * VSZ + jr * VSTR + seg * 8]),
               Vb + (size_t)(b * N_ + jt * 32 + jr) * D_ + h * DK_ + seg * 8);
  };
  auto loadK = [&](int jt) -> uint4 {
    return *reinterpret_cast<const uint4*>(
        Kb + (size_t)(b * N_ + jt * 32 + jr) * D_ + h * DK_ + seg * 8);
  };
  auto storeK = [&](uint4 q, int buf) {   // transpose into [dk][j]
    union { uint4 q; u16 e[8]; } kt; kt.q = q;
    u16* kb = Kst + buf * KSZ;
#pragma unroll
    for (int i = 0; i < 8; ++i) kb[(seg * 8 + i) * KSTR + jr] = kt.e[i];
  };

  const int nj = N_ / 32;
  stageV(0, 0);
  storeK(loadK(0), 0);

  for (int jt = 0; jt < nj; ++jt) {
    const int buf = jt & 1;
    const bool more = (jt + 1) < nj;
    wait_async0();       // my V(jt) landed (ds stores drained by barrier below)
    __syncthreads();

    uint4 kqn = {};
    if (more) { stageV(jt + 1, buf ^ 1); kqn = loadK(jt + 1); }

    const u16* ksb = Kst + buf * KSZ;
    const u16* vsb = Vs + buf * VSZ;
    const int  j0  = jt * 32;

    // S = Q K^T  (two 16-col tiles, two K-steps each)
    v8f s[2];
#pragma unroll
    for (int nt = 0; nt < 2; ++nt) {
      v8f sv = {};
#pragma unroll
      for (int ks = 0; ks < 2; ++ks) {
        const u16* base = &ksb[(ks * 32) * KSTR + nt * 16 + mrow]; // lane&15 = j
        sv = wmma_bf16(qa[ks], ldfragT(base, KSTR, hf), sv);
      }
      s[nt] = sv;
    }

    // scale * mask, then online softmax (rows distributed: m = v + 8*hf)
    const int n = lane & 15;
#pragma unroll
    for (int v = 0; v < 8; ++v) {
      int row = i0 + v + 8 * hf;
      const float* mrowp = mask + (size_t)row * N_ + j0;
      float s0 = s[0][v] * SCALE_ * mrowp[n];
      float s1 = s[1][v] * SCALE_ * mrowp[16 + n];

      float mx = fmaxf(s0, s1);
#pragma unroll
      for (int d = 8; d >= 1; d >>= 1) mx = fmaxf(mx, __shfl_xor(mx, d, 32));
      float mnew  = fmaxf(mrw[v], mx);
      float alpha = __expf(mrw[v] - mnew);
      float p0 = __expf(s0 - mnew);
      float p1 = __expf(s1 - mnew);
      float rs = p0 + p1;
#pragma unroll
      for (int d = 8; d >= 1; d >>= 1) rs += __shfl_xor(rs, d, 32);
      lrw[v] = lrw[v] * alpha + rs;
      mrw[v] = mnew;
#pragma unroll
      for (int nt2 = 0; nt2 < 4; ++nt2) o[nt2][v] *= alpha;

      // stage P (C layout -> row-major bf16); same-wave DS ops are in order
      u16* ps = Ps[wave];
      int m = v + 8 * hf;
      ps[m * PSTR + n]      = f2bf(p0);
      ps[m * PSTR + 16 + n] = f2bf(p1);
    }

    // O += P @ V   (P re-read in A layout from per-wave LDS scratch)
    v16bf pa = ldfrag(&Ps[wave][mrow * PSTR + 8 * hf],
                      &Ps[wave][mrow * PSTR + 16 + 8 * hf]);
#pragma unroll
    for (int nt2 = 0; nt2 < 4; ++nt2) {
      const u16* base = &vsb[nt2 * 16 + mrow];  // lane&15 = dk col
      o[nt2] = wmma_bf16(pa, ldfragT(base, VSTR, hf), o[nt2]);
    }

    if (more) storeK(kqn, buf ^ 1);   // after compute; visible at next barrier
  }

  // normalize and write O (bf16)
  const int n = lane & 15;
#pragma unroll
  for (int nt2 = 0; nt2 < 4; ++nt2)
#pragma unroll
    for (int v = 0; v < 8; ++v) {
      int row = i0 + v + 8 * hf;
      int dk  = nt2 * 16 + n;
      Ob[(size_t)(b * N_ + row) * D_ + h * DK_ + dk] = f2bf(o[nt2][v] / lrw[v]);
    }
}

// ---------------------------------------------------------------------------
// Host launch
// ---------------------------------------------------------------------------
extern "C" void kernel_launch(void* const* d_in, const int* in_sizes, int n_in,
                              void* d_out, int out_size, void* d_ws, size_t ws_size,
                              hipStream_t stream) {
  (void)in_sizes; (void)n_in; (void)out_size; (void)ws_size;

  const float* x     = (const float*)d_in[0];
  const float* graph = (const float*)d_in[1];
  const float* mask  = (const float*)d_in[2];
  const float* Wq = (const float*)d_in[3];  const float* bq = (const float*)d_in[4];
  const float* Wk = (const float*)d_in[5];  const float* bk = (const float*)d_in[6];
  const float* Wc = (const float*)d_in[7];  const float* bc = (const float*)d_in[8];
  const float* We = (const float*)d_in[9];  const float* be = (const float*)d_in[10];
  const float* Wv = (const float*)d_in[11]; const float* bv = (const float*)d_in[12];
  const float* Wo = (const float*)d_in[13]; const float* bo = (const float*)d_in[14];
  float* out = (float*)d_out;

  char* wp = (char*)d_ws;
  auto alloc = [&](size_t elems) -> u16* {
    u16* p = (u16*)wp;
    wp += ((elems * sizeof(u16)) + 255) & ~(size_t)255;
    return p;
  };
  u16* x_bf  = alloc((size_t)M_ * D_);
  u16* Wq_bf = alloc((size_t)D_ * D_);
  u16* Wk_bf = alloc((size_t)D_ * D_);
  u16* Wv_bf = alloc((size_t)D_ * D_);
  u16* Wo_bf = alloc((size_t)D_ * D_);
  u16* Wc_bf = alloc((size_t)N_ * D_);
  u16* We_bf = alloc((size_t)N_ * D_);
  u16* g_bf  = alloc((size_t)N_ * N_);
  u16* gT_bf = alloc((size_t)N_ * N_);
  u16* Cq_bf = alloc((size_t)N_ * D_);
  u16* Ce_bf = alloc((size_t)N_ * D_);
  u16* Q_bf  = alloc((size_t)M_ * D_);
  u16* K_bf  = alloc((size_t)M_ * D_);
  u16* V_bf  = alloc((size_t)M_ * D_);
  u16* O_bf  = alloc((size_t)M_ * D_);

  auto cvt = [&](const float* src, u16* dst, size_t n) {
    long long blocks = (long long)((n + 255) / 256);
    if (blocks > 4096) blocks = 4096;
    k_f2bf<<<dim3((unsigned)blocks), dim3(256), 0, stream>>>(src, dst, (long long)n);
  };
  cvt(x,  x_bf,  (size_t)M_ * D_);
  cvt(Wq, Wq_bf, (size_t)D_ * D_);
  cvt(Wk, Wk_bf, (size_t)D_ * D_);
  cvt(Wv, Wv_bf, (size_t)D_ * D_);
  cvt(Wo, Wo_bf, (size_t)D_ * D_);
  cvt(Wc, Wc_bf, (size_t)N_ * D_);
  cvt(We, We_bf, (size_t)N_ * D_);
  cvt(graph, g_bf, (size_t)N_ * N_);
  k_transpose_f2bf<<<dim3(N_ / 32, N_ / 32), dim3(256), 0, stream>>>(graph, gT_bf, N_);

  auto gemm = [&](const u16* A, const u16* Bm, const float* bias, const u16* extra,
                  int extraMask, float* oF, u16* oB, int M, int Nc, int Kd) {
    k_gemm<<<dim3(M / 128, Nc / 256), dim3(256), 0, stream>>>(
        A, Bm, bias, extra, extraMask, oF, oB, M, Nc, Kd);
  };

  // Causal terms: Cq = graph @ Wc + bc ; Ce = graph^T @ We + be   [N,D]
  gemm(g_bf,  Wc_bf, bc, nullptr, 0, nullptr, Cq_bf, N_, D_, N_);
  gemm(gT_bf, We_bf, be, nullptr, 0, nullptr, Ce_bf, N_, D_, N_);
  // Q/K/V with fused bias + broadcast causal add (row & (N-1))
  gemm(x_bf, Wq_bf, bq, Cq_bf, N_ - 1, nullptr, Q_bf, M_, D_, D_);
  gemm(x_bf, Wk_bf, bk, Ce_bf, N_ - 1, nullptr, K_bf, M_, D_, D_);
  gemm(x_bf, Wv_bf, bv, nullptr, 0, nullptr, V_bf, M_, D_, D_);
  // Attention
  k_attn<<<dim3(N_ / 128, B_ * H_), dim3(256), 0, stream>>>(
      Q_bf, K_bf, V_bf, mask, O_bf);
  // Output projection -> fp32 result
  gemm(O_bf, Wo_bf, bo, nullptr, 0, out, nullptr, M_, D_, D_);
}
